// InfiniTransformer_21895743275674
// MI455X (gfx1250) — compile-verified, matrix operand order checked by reading, not run on hardware
//
#include <hip/hip_runtime.h>
#include <math.h>

// B=2, S=8192, D=1024, H=8, Dk=Dv=128, L=512, DH=4096, nseg=16
typedef __attribute__((ext_vector_type(16))) __bf16 v16bf;
typedef __attribute__((ext_vector_type(4)))  __bf16 v4bf;
typedef __attribute__((ext_vector_type(8)))  float  v8f;
typedef __attribute__((ext_vector_type(4)))  int    v4i;

#define WMMA(a, b, c) \
  __builtin_amdgcn_wmma_f32_16x16x32_bf16(false, (a), false, (b), (short)0, (c), false, false)

#define LDT 40    // bf16 row stride for 32-wide K tiles (+8 pad; conflict-free)
#define MLD 136   // bf16 row stride for 128-wide mem tiles (+8 pad)

#if __has_builtin(__builtin_amdgcn_global_load_async_to_lds_b128) && \
    __has_builtin(__builtin_amdgcn_s_wait_asynccnt)
#define USE_ASYNC 1
#else
#define USE_ASYNC 0
#endif

// Copy 16 bytes global->LDS (async path uses GLOBAL_LOAD_ASYNC_TO_LDS_B128,
// tracked by ASYNCcnt; fallback stages through VGPRs).
__device__ __forceinline__ void cp16(const __bf16* g, __bf16* l) {
#if USE_ASYNC
  __builtin_amdgcn_global_load_async_to_lds_b128(
      (__attribute__((address_space(1))) v4i*)g,
      (__attribute__((address_space(3))) v4i*)l, 0, 0);
#else
  *(int4*)l = *(const int4*)g;
#endif
}

// A/B fragment (16x32 bf16): lane-half `sel` selects K runs [8s,8s+8) and
// [16+8s,16+8s+8) of a contiguous row (ISA 16-bit A-matrix layout).
__device__ __forceinline__ v16bf frag_row(const __bf16* row, int sel) {
  v16bf f;
#pragma unroll
  for (int i = 0; i < 8; ++i) f[i] = row[8 * sel + i];
#pragma unroll
  for (int i = 0; i < 8; ++i) f[8 + i] = row[16 + 8 * sel + i];
  return f;
}

// ---------------------------------------------------------------------------
// One-time conversions: f32 -> bf16 elementwise, and W[K,N] f32 -> Wt[N,K] bf16.
// ---------------------------------------------------------------------------
__global__ __launch_bounds__(256) void cvt_kernel(const float* __restrict__ in,
                                                  __bf16* __restrict__ out,
                                                  int n4) {
  const int i = blockIdx.x * 256 + threadIdx.x;
  if (i < n4) {
    const float4 f = ((const float4*)in)[i];
    v4bf o;
    o[0] = (__bf16)f.x; o[1] = (__bf16)f.y; o[2] = (__bf16)f.z; o[3] = (__bf16)f.w;
    ((v4bf*)out)[i] = o;
  }
}

__global__ __launch_bounds__(256) void transpose_cvt_kernel(
    const float* __restrict__ W, __bf16* __restrict__ Wt, int K, int N) {
  __shared__ float tile[32][33];
  const int k0 = blockIdx.y * 32, n0 = blockIdx.x * 32;
  const int tx = threadIdx.x & 31, ty = threadIdx.x >> 5;  // 32 x 8
#pragma unroll
  for (int r = 0; r < 32; r += 8)
    tile[ty + r][tx] = W[(size_t)(k0 + ty + r) * N + n0 + tx];
  __syncthreads();
#pragma unroll
  for (int r = 0; r < 32; r += 8)
    Wt[(size_t)(n0 + ty + r) * K + k0 + tx] = (__bf16)tile[tx][ty + r];
}

// ---------------------------------------------------------------------------
// GEMM: C[M,N] = act(A[M,K] @ Wt[N,K]^T + bias) (+ resid). All-bf16 operands,
// f32 WMMA accumulation. 128x128x32 block tile, 8 waves (4x2), double-buffered
// LDS with async global->LDS staging, one barrier per K-step.
// ---------------------------------------------------------------------------
__global__ __launch_bounds__(256) void gemm_kernel(
    const __bf16* __restrict__ A, const __bf16* __restrict__ Wt,
    const float* __restrict__ bias, const float* __restrict__ resid,
    void* __restrict__ Cout, int M, int N, int K, int act, int out_bf16) {
  __shared__ __bf16 As[2][128 * LDT];
  __shared__ __bf16 Bs[2][128 * LDT];

  const int tid = threadIdx.x;
  const int wave = tid >> 5, lane = tid & 31;
  const int lh = lane >> 4, ln = lane & 15;
  const int bm = blockIdx.y * 128, bn = blockIdx.x * 128;
  const int m0 = (wave >> 1) * 32, n0 = (wave & 1) * 64;

  v8f acc[2][4] = {};

  // Staging map: thread -> (row r, 32B chunk c) of the 128x32 bf16 tile.
  const int r = tid >> 1, c = (tid & 1) * 16;
  const __bf16* gA = A + (size_t)(bm + r) * K + c;
  const __bf16* gB = Wt + (size_t)(bn + r) * K + c;
  __bf16* lA = &As[0][r * LDT + c];
  __bf16* lB = &Bs[0][r * LDT + c];
  const int bufstride = 128 * LDT;

  // Prologue: stage K-step 0 into buffer 0.
  cp16(gA, lA); cp16(gA + 8, lA + 8);
  cp16(gB, lB); cp16(gB + 8, lB + 8);

  const int nk = K >> 5;
  for (int ks = 0; ks < nk; ++ks) {
    const int cur = ks & 1;
#if USE_ASYNC
    __builtin_amdgcn_s_wait_asynccnt(0);
#endif
    __syncthreads();
    if (ks + 1 < nk) {
      const int nxt = cur ^ 1;
      const int ko = (ks + 1) << 5;
      cp16(gA + ko, lA + nxt * bufstride);
      cp16(gA + ko + 8, lA + nxt * bufstride + 8);
      cp16(gB + ko, lB + nxt * bufstride);
      cp16(gB + ko + 8, lB + nxt * bufstride + 8);
#if !USE_ASYNC
      __builtin_prefetch(gA + ko + 32, 0, 1);
#endif
    }
#pragma unroll
    for (int mt = 0; mt < 2; ++mt) {
      v16bf af = frag_row(&As[cur][(m0 + mt * 16 + ln) * LDT], lh);
#pragma unroll
      for (int nt = 0; nt < 4; ++nt) {
        v16bf bf = frag_row(&Bs[cur][(n0 + nt * 16 + ln) * LDT], lh);
        acc[mt][nt] = WMMA(af, bf, acc[mt][nt]);
      }
    }
  }

#pragma unroll
  for (int mt = 0; mt < 2; ++mt)
#pragma unroll
    for (int nt = 0; nt < 4; ++nt) {
      const int col = bn + n0 + nt * 16 + ln;
      const float bvv = bias ? bias[col] : 0.f;
#pragma unroll
      for (int rr = 0; rr < 8; ++rr) {
        const int row = bm + m0 + mt * 16 + rr + 8 * lh;
        float vv = acc[mt][nt][rr] + bvv;
        if (act == 1) vv = 0.5f * vv * (1.f + erff(vv * 0.70710678118654752f));
        if (resid) vv += resid[(size_t)row * N + col];
        if (out_bf16)
          ((__bf16*)Cout)[(size_t)row * N + col] = (__bf16)vv;
        else
          ((float*)Cout)[(size_t)row * N + col] = vv;
      }
    }
}

// ---------------------------------------------------------------------------
// Pass 1: per-segment memory sums Mseg = (elu(k)+1)^T @ v (128x128) and
// Zseg = column sums of elu(k)+1. One block per (b,h,seg).
// ---------------------------------------------------------------------------
__global__ __launch_bounds__(256) void segmem_kernel(
    const __bf16* __restrict__ kbuf, const __bf16* __restrict__ vbuf,
    float* __restrict__ Mseg, float* __restrict__ Zseg) {
  __shared__ __bf16 sA[128 * LDT];  // sk^T chunk [dk][pos]
  __shared__ __bf16 sB[128 * LDT];  // v^T  chunk [dv][pos]
  __shared__ float zsh[128];

  const int tid = threadIdx.x;
  const int wave = tid >> 5, lane = tid & 31;
  const int lh = lane >> 4, ln = lane & 15;
  const int blk = blockIdx.x;  // (b*8+h)*16 + seg
  const int seg = blk & 15, bh = blk >> 4;
  const int h = bh & 7, b = bh >> 3;
  const size_t base = ((size_t)b * 8192 + seg * 512) * 1024 + h * 128;

  if (tid < 128) zsh[tid] = 0.f;
  v8f acc[8] = {};
  float zp[16];
#pragma unroll
  for (int i = 0; i < 16; ++i) zp[i] = 0.f;

  const int psub = tid >> 3, dbase = (tid & 7) * 16;
  for (int cix = 0; cix < 16; ++cix) {
    const __bf16* kp = kbuf + base + (size_t)(cix * 32 + psub) * 1024 + dbase;
    const __bf16* vp = vbuf + base + (size_t)(cix * 32 + psub) * 1024 + dbase;
#pragma unroll
    for (int i = 0; i < 16; ++i) {
      const float kv = (float)kp[i];
      const float sk = kv > 0.f ? kv + 1.f : __expf(kv);  // elu(k)+1
      zp[i] += sk;
      sA[(dbase + i) * LDT + psub] = (__bf16)sk;
      sB[(dbase + i) * LDT + psub] = vp[i];
    }
    __syncthreads();
    v16bf af = frag_row(&sA[(wave * 16 + ln) * LDT], lh);
#pragma unroll
    for (int nt = 0; nt < 8; ++nt) {
      v16bf bf = frag_row(&sB[(nt * 16 + ln) * LDT], lh);
      acc[nt] = WMMA(af, bf, acc[nt]);
    }
    __syncthreads();
  }
  float* mo = Mseg + (size_t)blk * 16384;
#pragma unroll
  for (int nt = 0; nt < 8; ++nt)
#pragma unroll
    for (int rr = 0; rr < 8; ++rr)
      mo[(wave * 16 + rr + 8 * lh) * 128 + nt * 16 + ln] = acc[nt][rr];
#pragma unroll
  for (int i = 0; i < 16; ++i) atomicAdd(&zsh[dbase + i], zp[i]);
  __syncthreads();
  if (tid < 128) Zseg[(size_t)blk * 128 + tid] = zsh[tid];
}

// ---------------------------------------------------------------------------
// Exclusive prefix over 16 segments (the scan's linear recurrence), f32.
// ---------------------------------------------------------------------------
__global__ __launch_bounds__(256) void prefix_kernel(
    const float* __restrict__ Mseg, const float* __restrict__ Zseg,
    float* __restrict__ Mpre, float* __restrict__ Zpre) {
  const int idx = blockIdx.x * 256 + threadIdx.x;  // 16 * 16384
  const int bh = idx >> 14, e = idx & 16383;
  float run = 0.f;  // mem0 = 0
  for (int s = 0; s < 16; ++s) {
    const size_t o = (size_t)(bh * 16 + s) * 16384 + e;
    Mpre[o] = run;
    run += Mseg[o];
  }
  if (e < 128) {
    float rz = 1.f / 128.f;  // z0 = 1/Dk
    for (int s = 0; s < 16; ++s) {
      const size_t o = (size_t)(bh * 16 + s) * 128 + e;
      Zpre[o] = rz;
      rz += Zseg[o];
    }
  }
}

// ---------------------------------------------------------------------------
// Pass 2: per (b,h,seg,q-tile of 128 rows): retrieval (sq@mem)/(sq.z) plus
// flash-style causal dot attention within the segment; gated combine.
// ---------------------------------------------------------------------------
__global__ __launch_bounds__(256) void attn_kernel(
    const __bf16* __restrict__ qbuf, const __bf16* __restrict__ kbuf,
    const __bf16* __restrict__ vbuf, const float* __restrict__ Mpre,
    const float* __restrict__ Zpre, const float* __restrict__ betas,
    __bf16* __restrict__ att) {
  __shared__ __bf16 memT[128 * MLD];  // mem^T bf16: [dv][dk]
  __shared__ __bf16 sP[128 * LDT];    // softmax P (per-wave 16x32 slices)
  __shared__ __bf16 sV[128 * LDT];    // v^T chunk: [dv][key]
  __shared__ float zS[128];
  __shared__ float denomS[128];

  const int tid = threadIdx.x;
  const int wave = tid >> 5, lane = tid & 31;
  const int lh = lane >> 4, ln = lane & 15;
  const int blk = blockIdx.x;  // ((b*8+h)*16 + seg)*4 + qt
  const int qt = blk & 3, seg = (blk >> 2) & 15, bh = blk >> 6;
  const int h = bh & 7, b = bh >> 3;
  const size_t segbase = ((size_t)b * 8192 + seg * 512) * 1024 + h * 128;

  const float* mp = Mpre + (size_t)(bh * 16 + seg) * 16384;
  for (int i = tid; i < 16384; i += 256)
    memT[(i & 127) * MLD + (i >> 7)] = (__bf16)mp[i];
  if (tid < 128) zS[tid] = Zpre[(size_t)(bh * 16 + seg) * 128 + tid];
  __syncthreads();

  if (tid < 128) {
    const __bf16* qp = qbuf + segbase + (size_t)(qt * 128 + tid) * 1024;
    float s = 0.f;
    for (int d = 0; d < 128; ++d) {
      const float x = (float)qp[d];
      s += (x > 0.f ? x + 1.f : __expf(x)) * zS[d];
    }
    denomS[tid] = s;
  }
  __syncthreads();

  const int rowloc = qt * 128 + wave * 16;
  const __bf16* qrow = qbuf + segbase + (size_t)(rowloc + ln) * 1024;

  v16bf aq[4], asq[4];
#pragma unroll
  for (int ds = 0; ds < 4; ++ds) {
    aq[ds] = frag_row(qrow + ds * 32, lh);
    v16bf g;
#pragma unroll
    for (int hf = 0; hf < 2; ++hf) {
      const __bf16* p = qrow + ds * 32 + 16 * hf + 8 * lh;
#pragma unroll
      for (int i = 0; i < 8; ++i) {
        const float x = (float)p[i];
        g[hf * 8 + i] = (__bf16)(x > 0.f ? x + 1.f : __expf(x));
      }
    }
    asq[ds] = g;
  }

  v8f macc[8] = {};
#pragma unroll
  for (int nt = 0; nt < 8; ++nt)
#pragma unroll
    for (int ds = 0; ds < 4; ++ds) {
      v16bf bf = frag_row(&memT[(nt * 16 + ln) * MLD + ds * 32], lh);
      macc[nt] = WMMA(asq[ds], bf, macc[nt]);
    }

  v8f oacc[8] = {};
  float mstat[8], lstat[8];
#pragma unroll
  for (int rr = 0; rr < 8; ++rr) { mstat[rr] = -3.0e38f; lstat[rr] = 0.f; }

  const float scale = 0.088388347648318447f;  // 1/sqrt(128)
  const int nchunks = (qt + 1) * 4;
  const int psub = tid >> 3, dbase = (tid & 7) * 16;

  for (int jc = 0; jc < nchunks; ++jc) {
    const __bf16* vp =
        vbuf + segbase + (size_t)(jc * 32 + psub) * 1024 + dbase;
#pragma unroll
    for (int i = 0; i < 16; ++i) sV[(dbase + i) * LDT + psub] = vp[i];
    __syncthreads();

    v8f sacc[2] = {};
#pragma unroll
    for (int ds = 0; ds < 4; ++ds)
#pragma unroll
      for (int t = 0; t < 2; ++t) {
        const __bf16* kp =
            kbuf + segbase + (size_t)(jc * 32 + t * 16 + ln) * 1024 + ds * 32;
        sacc[t] = WMMA(aq[ds], frag_row(kp, lh), sacc[t]);
      }

    const bool masked = (jc >= qt * 4);
#pragma unroll
    for (int rr = 0; rr < 8; ++rr) {
      const int qpos = rowloc + rr + 8 * lh;
      float s0 = sacc[0][rr] * scale;
      float s1 = sacc[1][rr] * scale;
      if (masked) {
        if (jc * 32 + ln > qpos) s0 = -1.0e30f;
        if (jc * 32 + 16 + ln > qpos) s1 = -1.0e30f;
      }
      float mx = fmaxf(s0, s1);
#pragma unroll
      for (int off = 1; off < 16; off <<= 1)
        mx = fmaxf(mx, __shfl_xor(mx, off, 32));  // within 16-lane half
      const float mold = mstat[rr];
      const float mnew = fmaxf(mold, mx);
      const float corr = __expf(mold - mnew);
      mstat[rr] = mnew;
      const float p0 = __expf(s0 - mnew);
      const float p1 = __expf(s1 - mnew);
      float ps = p0 + p1;
#pragma unroll
      for (int off = 1; off < 16; off <<= 1) ps += __shfl_xor(ps, off, 32);
      lstat[rr] = lstat[rr] * corr + ps;
#pragma unroll
      for (int nt = 0; nt < 8; ++nt) oacc[nt][rr] *= corr;
      sP[(wave * 16 + rr + 8 * lh) * LDT + ln] = (__bf16)p0;
      sP[(wave * 16 + rr + 8 * lh) * LDT + 16 + ln] = (__bf16)p1;
    }
    __syncthreads();

    v16bf pf = frag_row(&sP[(wave * 16 + ln) * LDT], lh);
#pragma unroll
    for (int nt = 0; nt < 8; ++nt) {
      v16bf vf = frag_row(&sV[(nt * 16 + ln) * LDT], lh);
      oacc[nt] = WMMA(pf, vf, oacc[nt]);
    }
    __syncthreads();
  }

#pragma unroll
  for (int nt = 0; nt < 8; ++nt) {
    const float gate = 1.f / (1.f + __expf(-betas[h * 128 + nt * 16 + ln]));
#pragma unroll
    for (int rr = 0; rr < 8; ++rr) {
      const int row = rowloc + rr + 8 * lh;
      const float amem = macc[nt][rr] / denomS[wave * 16 + rr + 8 * lh];
      const float adot = oacc[nt][rr] / lstat[rr];
      att[segbase + (size_t)row * 1024 + nt * 16 + ln] =
          (__bf16)(gate * amem + (1.f - gate) * adot);
    }
  }
}

// ---------------------------------------------------------------------------
// In-place LayerNorm over rows of 1024 (residual fused in final GEMM).
// ---------------------------------------------------------------------------
__global__ __launch_bounds__(256) void ln_kernel(
    float* __restrict__ y, const float* __restrict__ g,
    const float* __restrict__ bta) {
  __shared__ float red[16];
  const int tid = threadIdx.x;
  float* p = y + (size_t)blockIdx.x * 1024;
  float x[4];
  float s = 0.f, s2 = 0.f;
#pragma unroll
  for (int i = 0; i < 4; ++i) {
    x[i] = p[tid + 256 * i];
    s += x[i];
    s2 += x[i] * x[i];
  }
#pragma unroll
  for (int off = 1; off < 32; off <<= 1) {
    s += __shfl_xor(s, off, 32);
    s2 += __shfl_xor(s2, off, 32);
  }
  if ((tid & 31) == 0) { red[tid >> 5] = s; red[8 + (tid >> 5)] = s2; }
  __syncthreads();
  if (tid == 0) {
    float ts = 0.f, ts2 = 0.f;
    for (int w = 0; w < 8; ++w) { ts += red[w]; ts2 += red[8 + w]; }
    const float mu = ts / 1024.f;
    const float var = ts2 / 1024.f - mu * mu;
    red[0] = mu;
    red[1] = rsqrtf(var + 1e-5f);
  }
  __syncthreads();
  const float mu = red[0], rstd = red[1];
#pragma unroll
  for (int i = 0; i < 4; ++i) {
    const int cc = tid + 256 * i;
    p[cc] = (x[i] - mu) * rstd * g[cc] + bta[cc];
  }
}

// ---------------------------------------------------------------------------
extern "C" void kernel_launch(void* const* d_in, const int* in_sizes, int n_in,
                              void* d_out, int out_size, void* d_ws,
                              size_t ws_size, hipStream_t stream) {
  (void)in_sizes; (void)n_in; (void)out_size; (void)ws_size;
  const float* x     = (const float*)d_in[0];
  const float* Wq    = (const float*)d_in[1];
  const float* bq    = (const float*)d_in[2];
  const float* Wk    = (const float*)d_in[3];
  const float* bk    = (const float*)d_in[4];
  const float* Wv    = (const float*)d_in[5];
  const float* bv    = (const float*)d_in[6];
  const float* betas = (const float*)d_in[7];
  const float* Wo    = (const float*)d_in[8];
  const float* bo    = (const float*)d_in[9];
  const float* W1    = (const float*)d_in[10];
  const float* b1    = (const float*)d_in[11];
  const float* W2    = (const float*)d_in[12];
  const float* b2    = (const float*)d_in[13];
  const float* lng   = (const float*)d_in[14];
  const float* lnb   = (const float*)d_in[15];
  float* out = (float*)d_out;

  // Workspace carve-up (deterministic bump allocator, 256B aligned).
  char* wp = (char*)d_ws;
  auto alloc = [&](size_t bytes) {
    char* rr = wp;
    wp += (bytes + 255) & ~(size_t)255;
    return rr;
  };
  const size_t MT = (size_t)16384 * 1024;  // tokens x 1024
  __bf16* xb   = (__bf16*)alloc(MT * 2);
  __bf16* qb   = (__bf16*)alloc(MT * 2);
  __bf16* kb   = (__bf16*)alloc(MT * 2);
  __bf16* vb   = (__bf16*)alloc(MT * 2);
  __bf16* attb = (__bf16*)alloc(MT * 2);
  __bf16* hb   = (__bf16*)alloc(MT * 4 * 2);  // tokens x 4096
  __bf16* Wqt  = (__bf16*)alloc((size_t)1024 * 1024 * 2);
  __bf16* Wkt  = (__bf16*)alloc((size_t)1024 * 1024 * 2);
  __bf16* Wvt  = (__bf16*)alloc((size_t)1024 * 1024 * 2);
  __bf16* Wot  = (__bf16*)alloc((size_t)1024 * 1024 * 2);
  __bf16* W1t  = (__bf16*)alloc((size_t)1024 * 4096 * 2);
  __bf16* W2t  = (__bf16*)alloc((size_t)4096 * 1024 * 2);
  float* Mseg  = (float*)alloc((size_t)256 * 16384 * 4);
  float* Mpre  = (float*)alloc((size_t)256 * 16384 * 4);
  float* Zseg  = (float*)alloc((size_t)256 * 128 * 4);
  float* Zpre  = (float*)alloc((size_t)256 * 128 * 4);
  __bf16* ab = qb;  // q dead after attn; reuse for Wo output

  const dim3 blk(256);
  const dim3 g1024(8, 128);
  const dim3 g4096(32, 128);

  // One-time bf16 conversion + weight transposes.
  cvt_kernel<<<16384, blk, 0, stream>>>(x, xb, (int)(MT / 4));
  transpose_cvt_kernel<<<dim3(32, 32), blk, 0, stream>>>(Wq, Wqt, 1024, 1024);
  transpose_cvt_kernel<<<dim3(32, 32), blk, 0, stream>>>(Wk, Wkt, 1024, 1024);
  transpose_cvt_kernel<<<dim3(32, 32), blk, 0, stream>>>(Wv, Wvt, 1024, 1024);
  transpose_cvt_kernel<<<dim3(32, 32), blk, 0, stream>>>(Wo, Wot, 1024, 1024);
  transpose_cvt_kernel<<<dim3(128, 32), blk, 0, stream>>>(W1, W1t, 1024, 4096);
  transpose_cvt_kernel<<<dim3(32, 128), blk, 0, stream>>>(W2, W2t, 4096, 1024);

  // QKV projections (bf16 out).
  gemm_kernel<<<g1024, blk, 0, stream>>>(xb, Wqt, bq, nullptr, qb, 16384, 1024, 1024, 0, 1);
  gemm_kernel<<<g1024, blk, 0, stream>>>(xb, Wkt, bk, nullptr, kb, 16384, 1024, 1024, 0, 1);
  gemm_kernel<<<g1024, blk, 0, stream>>>(xb, Wvt, bv, nullptr, vb, 16384, 1024, 1024, 0, 1);

  // Compressive-memory scan (parallel segment sums + tiny prefix) + attention.
  segmem_kernel<<<256, blk, 0, stream>>>(kb, vb, Mseg, Zseg);
  prefix_kernel<<<1024, blk, 0, stream>>>(Mseg, Zseg, Mpre, Zpre);
  attn_kernel<<<1024, blk, 0, stream>>>(qb, kb, vb, Mpre, Zpre, betas, attb);

  // Output projection + FFN (+ fused residual) + LN.
  gemm_kernel<<<g1024, blk, 0, stream>>>(attb, Wot, bo, nullptr, ab, 16384, 1024, 1024, 0, 1);
  gemm_kernel<<<g4096, blk, 0, stream>>>(ab, W1t, b1, nullptr, hb, 16384, 4096, 1024, 1, 1);
  gemm_kernel<<<g1024, blk, 0, stream>>>(hb, W2t, b2, x, out, 16384, 1024, 4096, 0, 0);
  ln_kernel<<<16384, blk, 0, stream>>>(out, lng, lnb);
}
